// CartesianDecomposedAttention_876173328871
// MI455X (gfx1250) — matrix operand order repaired and checked by reading, not compiled, not run on hardware
//
#include <hip/hip_runtime.h>

// ---------------------------------------------------------------------------
// CartesianDecomposedAttention for MI455X (gfx1250, wave32, WMMA).
// Pipeline: f32->bf16 cvt -> complex GEMM (QKV) -> RoPE/head-split ->
//           fused attention (256KB LDS score tiles) -> complex GEMM (out proj)
// Matmuls via v_wmma_f32_16x16x32_bf16 (f32 accumulate); GEMM tiles staged
// with async global->LDS DMA (ASYNCcnt) and double buffering.
// ---------------------------------------------------------------------------

typedef __bf16        bf16x16 __attribute__((ext_vector_type(16)));
typedef float         f32x8   __attribute__((ext_vector_type(8)));
typedef unsigned int  u32x8   __attribute__((ext_vector_type(8)));

__device__ __forceinline__ unsigned short f2bf(float f) {
  unsigned u = __builtin_bit_cast(unsigned, f);
  u += 0x7FFFu + ((u >> 16) & 1u);          // round-to-nearest-even
  return (unsigned short)(u >> 16);
}

__device__ __forceinline__ f32x8 wmma_bf16(bf16x16 a, bf16x16 b, f32x8 c) {
  // (neg_a, A, neg_b, B, c_mod, C, reuse_a, reuse_b)
  return __builtin_amdgcn_wmma_f32_16x16x32_bf16(false, a, false, b,
                                                 (short)0, c, false, false);
}

__device__ __forceinline__ bf16x16 make_frag(uint4 lo, uint4 hi) {
  u32x8 r;
  r[0] = lo.x; r[1] = lo.y; r[2] = lo.z; r[3] = lo.w;
  r[4] = hi.x; r[5] = hi.y; r[6] = hi.z; r[7] = hi.w;
  return __builtin_bit_cast(bf16x16, r);
}

__device__ __forceinline__ bf16x16 neg_frag(bf16x16 f) {
  u32x8 r = __builtin_bit_cast(u32x8, f);
  r ^= 0x80008000u;                          // flip bf16 sign bits (packed pair)
  return __builtin_bit_cast(bf16x16, f = __builtin_bit_cast(bf16x16, r));
}

// A-matrix 16x32 bf16 fragment (ISA 7.12.2): lane m=lane&15;
// chunk0 at K = (lane<16 ? 0 : 8), chunk1 at +16 elements.
__device__ __forceinline__ bf16x16 load_frag_a(const unsigned short* p,
                                               int rowStride, int lane) {
  int m    = lane & 15;
  int koff = (lane < 16) ? 0 : 8;
  const uint4* q = (const uint4*)(p + m * rowStride + koff);
  return make_frag(q[0], q[2]);              // +16 elems = +2 uint4
}

// B-matrix 32x16 bf16 fragment: lane n=lane&15;
// chunk0 at K = (lane<16 ? 0 : 8), chunk1 at +16 elements.
__device__ __forceinline__ bf16x16 load_frag_b(const unsigned short* p,
                                               int rowStride, int lane) {
  int n  = lane & 15;
  int kb = (lane < 16) ? 0 : 8;
  const uint4* q = (const uint4*)(p + n * rowStride + kb);
  return make_frag(q[0], q[2]);              // +16 elems = +2 uint4
}

// Async DMA: 16 bytes global -> LDS, tracked by ASYNCcnt.
// LDS aperture keeps the LDS byte offset in addr[31:0], so truncating the
// flat pointer yields the ds address operand.
__device__ __forceinline__ void async_b128(unsigned short* ldst,
                                           const unsigned short* gsrc) {
  unsigned lds = (unsigned)(unsigned long long)ldst;
  unsigned long long ga = (unsigned long long)gsrc;
  asm volatile("global_load_async_to_lds_b128 %0, %1, off"
               :: "v"(lds), "v"(ga) : "memory");
}

// ---------------------------------------------------------------------------
// fp32 -> bf16 conversion
// ---------------------------------------------------------------------------
__global__ __launch_bounds__(256) void cvt_f32_bf16(const float* __restrict__ src,
                                                    unsigned short* __restrict__ dst,
                                                    int n) {
  int i = blockIdx.x * 256 + threadIdx.x;
  if (i < n) dst[i] = f2bf(src[i]);
}

// ---------------------------------------------------------------------------
// Complex GEMM: C = A * W^T  (complex, no conjugation), A:[M,K], W:[N,K]
//   Cr = Ar*Wr^T - Ai*Wi^T ; Ci = Ar*Wi^T + Ai*Wr^T
// Workgroup tile 128(M) x 64(N); 8 waves as 4x2; each wave: 2x2 WMMA tiles.
// LDS staged with double-buffered async global->LDS DMA.
// ---------------------------------------------------------------------------
#define TM 128
#define TN 64
#define TK 32

__global__ __launch_bounds__(256) void cgemm_cplx_bf16(
    const unsigned short* __restrict__ Ar, const unsigned short* __restrict__ Ai,
    const unsigned short* __restrict__ Wr, const unsigned short* __restrict__ Wi,
    float* __restrict__ Cr, float* __restrict__ Ci, int M, int N, int K) {
  __shared__ unsigned short sA[2][2][TM * TK];   // [buf][re/im]
  __shared__ unsigned short sW[2][2][TN * TK];   // [buf][re/im]

  const int tid  = threadIdx.x;
  const int lane = tid & 31;
  const int wave = tid >> 5;
  const int m0 = blockIdx.y * TM;
  const int n0 = blockIdx.x * TN;
  const int mw = (wave & 3) * 32;
  const int nw = (wave >> 2) * 32;

  const f32x8 zero = {0.f, 0.f, 0.f, 0.f, 0.f, 0.f, 0.f, 0.f};
  f32x8 aR[2][2], aI[2][2];
#pragma unroll
  for (int i = 0; i < 2; ++i)
#pragma unroll
    for (int j = 0; j < 2; ++j) { aR[i][j] = zero; aI[i][j] = zero; }

  // cooperative-load mapping: A 128x32 (16 elems/thread), W 64x32 (8/thread)
  const int arow = tid >> 1, acol = (tid & 1) * 16;
  const int brow = tid >> 2, bcol = (tid & 3) * 8;

  // issue 6 async b128 DMAs filling buffer `buf` with K-chunk at k0
  auto fill = [&](int buf, int k0) {
    const unsigned short* gAr = Ar + (size_t)(m0 + arow) * K + k0 + acol;
    const unsigned short* gAi = Ai + (size_t)(m0 + arow) * K + k0 + acol;
    unsigned short* dAr = &sA[buf][0][arow * TK + acol];
    unsigned short* dAi = &sA[buf][1][arow * TK + acol];
    async_b128(dAr,     gAr);
    async_b128(dAr + 8, gAr + 8);
    async_b128(dAi,     gAi);
    async_b128(dAi + 8, gAi + 8);
    async_b128(&sW[buf][0][brow * TK + bcol], Wr + (size_t)(n0 + brow) * K + k0 + bcol);
    async_b128(&sW[buf][1][brow * TK + bcol], Wi + (size_t)(n0 + brow) * K + k0 + bcol);
  };

  fill(0, 0);
  int cur = 0;
  for (int k0 = 0; k0 < K; k0 += TK) {
    if (k0 + TK < K) {
      fill(cur ^ 1, k0 + TK);                 // prefetch next chunk via DMA
      // 6 newest (next buffer) may stay in flight; older 6 must have landed
      asm volatile("s_wait_asynccnt 0x6" ::: "memory");
    } else {
      asm volatile("s_wait_asynccnt 0x0" ::: "memory");
    }
    __syncthreads();

    bf16x16 fAr[2], fAi[2], fAiN[2], fBr[2], fBi[2];
#pragma unroll
    for (int mi = 0; mi < 2; ++mi) {
      fAr[mi]  = load_frag_a(&sA[cur][0][(mw + mi * 16) * TK], TK, lane);
      fAi[mi]  = load_frag_a(&sA[cur][1][(mw + mi * 16) * TK], TK, lane);
      fAiN[mi] = neg_frag(fAi[mi]);
    }
#pragma unroll
    for (int ni = 0; ni < 2; ++ni) {
      fBr[ni] = load_frag_b(&sW[cur][0][(nw + ni * 16) * TK], TK, lane);
      fBi[ni] = load_frag_b(&sW[cur][1][(nw + ni * 16) * TK], TK, lane);
    }
#pragma unroll
    for (int mi = 0; mi < 2; ++mi)
#pragma unroll
      for (int ni = 0; ni < 2; ++ni) {
        aR[mi][ni] = wmma_bf16(fAr[mi],  fBr[ni], aR[mi][ni]);
        aR[mi][ni] = wmma_bf16(fAiN[mi], fBi[ni], aR[mi][ni]);  // - Ai*Wi
        aI[mi][ni] = wmma_bf16(fAr[mi],  fBi[ni], aI[mi][ni]);
        aI[mi][ni] = wmma_bf16(fAi[mi],  fBr[ni], aI[mi][ni]);
      }
    __syncthreads();   // all waves done with `cur` before it is refilled
    cur ^= 1;
  }

  const int col = lane & 15;
  const int rhi = (lane >> 4) << 3;
#pragma unroll
  for (int mi = 0; mi < 2; ++mi)
#pragma unroll
    for (int ni = 0; ni < 2; ++ni)
#pragma unroll
      for (int r = 0; r < 8; ++r) {
        size_t off = (size_t)(m0 + mw + mi * 16 + r + rhi) * N +
                     (n0 + nw + ni * 16 + col);
        Cr[off] = aR[mi][ni][r];
        Ci[off] = aI[mi][ni][r];
      }
}

// ---------------------------------------------------------------------------
// RoPE + head split. qkv:[2048,3072] f32 -> Q/K rotated bf16 [BH,512,64],
// V transposed bf16 [BH,64,512].
// ---------------------------------------------------------------------------
__global__ __launch_bounds__(256) void rope_split_kernel(
    const float* __restrict__ qkv_r, const float* __restrict__ qkv_i,
    unsigned short* __restrict__ Qr, unsigned short* __restrict__ Qi,
    unsigned short* __restrict__ Kr, unsigned short* __restrict__ Ki,
    unsigned short* __restrict__ Vtr, unsigned short* __restrict__ Vti) {
  int t  = blockIdx.x * 256 + threadIdx.x;   // 0 .. 2M-1
  int d  = t & 63;
  int s  = (t >> 6) & 511;
  int bh = t >> 15;                           // b*16 + h
  int h  = bh & 15, b = bh >> 4;
  size_t row = (size_t)(b * 512 + s) * 3072;
  int cq = h * 64 + d;
  float qr = qkv_r[row + cq],        qi = qkv_i[row + cq];
  float kr = qkv_r[row + 1024 + cq], ki = qkv_i[row + 1024 + cq];
  float vr = qkv_r[row + 2048 + cq], vi = qkv_i[row + 2048 + cq];
  // inv_freq = 10000^(-d/64) = exp(-d * ln(10000)/64)
  float ang = (float)s * __expf(-(float)d * 0.14391156f);
  float sn, cs; __sincosf(ang, &sn, &cs);
  size_t o = (size_t)bh * 32768 + (size_t)s * 64 + d;
  Qr[o] = f2bf(qr * cs - qi * sn);  Qi[o] = f2bf(qr * sn + qi * cs);
  Kr[o] = f2bf(kr * cs - ki * sn);  Ki[o] = f2bf(kr * sn + ki * cs);
  size_t ov = (size_t)bh * 32768 + (size_t)d * 512 + s;
  Vtr[ov] = f2bf(vr);  Vti[ov] = f2bf(vi);
}

// ---------------------------------------------------------------------------
// Fused attention per (b,h,q-tile of 64). Dynamic LDS = 256KB:
//   [0,128K)   sc_re f32 [64][512]   (later: awc bf16 [64][512] in [0,64K))
//   [128,256K) sc_im f32 [64][512]   (later: aws bf16 [64][512] in [128,192K))
// ---------------------------------------------------------------------------
__global__ __launch_bounds__(256) void attn_kernel(
    const unsigned short* __restrict__ Qr,  const unsigned short* __restrict__ Qi,
    const unsigned short* __restrict__ Kr,  const unsigned short* __restrict__ Ki,
    const unsigned short* __restrict__ Vtr, const unsigned short* __restrict__ Vti,
    unsigned short* __restrict__ Or, unsigned short* __restrict__ Oi) {
  extern __shared__ char smem[];
  float* sc_re = (float*)smem;
  float* sc_im = (float*)(smem + 64 * 512 * 4);
  unsigned short* awc = (unsigned short*)smem;
  unsigned short* aws = (unsigned short*)(smem + 64 * 512 * 4);

  const int tid  = threadIdx.x;
  const int lane = tid & 31;
  const int wave = tid >> 5;
  const int b = blockIdx.z, h = blockIdx.y, bh = b * 16 + h;
  const int q0 = blockIdx.x * 64;
  const float scale = 0.125f;                // 1/sqrt(64)

  const unsigned short* Qrb  = Qr  + (size_t)bh * 32768;
  const unsigned short* Qib  = Qi  + (size_t)bh * 32768;
  const unsigned short* Krb  = Kr  + (size_t)bh * 32768;
  const unsigned short* Kib  = Ki  + (size_t)bh * 32768;
  const unsigned short* Vtrb = Vtr + (size_t)bh * 32768;
  const unsigned short* Vtib = Vti + (size_t)bh * 32768;

  const f32x8 zero = {0.f, 0.f, 0.f, 0.f, 0.f, 0.f, 0.f, 0.f};
  const int qsub = wave & 3;                 // q sub-tile (16 rows)
  const int half = wave >> 2;                // key half (256 keys)

  // ---- phase 1: re/im scores via WMMA -------------------------------------
  bf16x16 fQr[2], fQi[2], fQrN[2];
#pragma unroll
  for (int c = 0; c < 2; ++c) {
    fQr[c]  = load_frag_a(Qrb + (size_t)(q0 + qsub * 16) * 64 + c * 32, 64, lane);
    fQi[c]  = load_frag_a(Qib + (size_t)(q0 + qsub * 16) * 64 + c * 32, 64, lane);
    fQrN[c] = neg_frag(fQr[c]);
  }
  for (int i = 0; i < 16; ++i) {
    int key0 = (half * 16 + i) * 16;
    f32x8 aR = zero, aI = zero;
#pragma unroll
    for (int c = 0; c < 2; ++c) {
      bf16x16 fKr = load_frag_b(Krb + (size_t)key0 * 64 + c * 32, 64, lane);
      bf16x16 fKi = load_frag_b(Kib + (size_t)key0 * 64 + c * 32, 64, lane);
      aR = wmma_bf16(fQr[c],  fKr, aR);      // Qr*Kr^T
      aR = wmma_bf16(fQi[c],  fKi, aR);      // + Qi*Ki^T
      aI = wmma_bf16(fQi[c],  fKr, aI);      // Qi*Kr^T
      aI = wmma_bf16(fQrN[c], fKi, aI);      // - Qr*Ki^T
    }
    int col = lane & 15, rhi = (lane >> 4) << 3;
#pragma unroll
    for (int r = 0; r < 8; ++r) {
      int row = qsub * 16 + r + rhi;
      sc_re[row * 512 + key0 + col] = aR[r] * scale;
      sc_im[row * 512 + key0 + col] = aI[r] * scale;
    }
  }
  __syncthreads();

  // ---- phase 2a: row softmax over sc_re (4 threads per row) ---------------
  {
    int r = tid >> 2, part = (tid & 3) * 128;
    float* row = sc_re + r * 512;
    float mx = -3.0e38f;
    for (int k = 0; k < 128; ++k) mx = fmaxf(mx, row[part + k]);
    mx = fmaxf(mx, __shfl_xor(mx, 1, 4));
    mx = fmaxf(mx, __shfl_xor(mx, 2, 4));
    float sum = 0.f;
    for (int k = 0; k < 128; ++k) {
      float e = __expf(row[part + k] - mx);
      row[part + k] = e;
      sum += e;
    }
    sum += __shfl_xor(sum, 1, 4);
    sum += __shfl_xor(sum, 2, 4);
    float inv = 1.f / sum;
    for (int k = 0; k < 128; ++k) row[part + k] *= inv;
  }
  __syncthreads();

  // ---- phase 2b: aw_cos = p*cos(im), aw_sin = p*sin(im), bf16 in place ----
  for (int j = 0; j < 8; ++j) {              // 8-row blocks avoid aliasing
    float pv[16], iv[16];
    int base = j * 4096 + tid * 16;
#pragma unroll
    for (int e = 0; e < 16; ++e) { pv[e] = sc_re[base + e]; iv[e] = sc_im[base + e]; }
    __syncthreads();
#pragma unroll
    for (int e = 0; e < 16; ++e) {
      float sn, cs; __sincosf(iv[e], &sn, &cs);
      awc[base + e] = f2bf(pv[e] * cs);
      aws[base + e] = f2bf(pv[e] * sn);
    }
    __syncthreads();
  }

  // ---- phase 3: out = (awc + i*aws) applied to V (complex) ----------------
  const int dpair = wave >> 2;               // this wave's pair of d-tiles
  f32x8 oR[2], oI[2];
  oR[0] = zero; oR[1] = zero; oI[0] = zero; oI[1] = zero;
  for (int kc = 0; kc < 16; ++kc) {          // 32-key chunks
    bf16x16 fC  = load_frag_a(awc + (qsub * 16) * 512 + kc * 32, 512, lane);
    bf16x16 fS  = load_frag_a(aws + (qsub * 16) * 512 + kc * 32, 512, lane);
    bf16x16 fSN = neg_frag(fS);
#pragma unroll
    for (int t = 0; t < 2; ++t) {
      int dt = dpair * 2 + t;
      bf16x16 fVr = load_frag_b(Vtrb + (size_t)(dt * 16) * 512 + kc * 32, 512, lane);
      bf16x16 fVi = load_frag_b(Vtib + (size_t)(dt * 16) * 512 + kc * 32, 512, lane);
      oR[t] = wmma_bf16(fC,  fVr, oR[t]);    // awc*Vr
      oR[t] = wmma_bf16(fSN, fVi, oR[t]);    // - aws*Vi
      oI[t] = wmma_bf16(fC,  fVi, oI[t]);    // awc*Vi
      oI[t] = wmma_bf16(fS,  fVr, oI[t]);    // + aws*Vr
    }
  }
  const int col = lane & 15, rhi = (lane >> 4) << 3;
#pragma unroll
  for (int t = 0; t < 2; ++t) {
    int dt = dpair * 2 + t;
#pragma unroll
    for (int r = 0; r < 8; ++r) {
      size_t row = (size_t)(b * 512 + q0 + qsub * 16 + r + rhi);
      int cc = h * 64 + dt * 16 + col;
      Or[row * 1024 + cc] = f2bf(oR[t][r]);
      Oi[row * 1024 + cc] = f2bf(oI[t][r]);
    }
  }
}

// ---------------------------------------------------------------------------
// Host-side launch
// ---------------------------------------------------------------------------
extern "C" void kernel_launch(void* const* d_in, const int* in_sizes, int n_in,
                              void* d_out, int out_size, void* d_ws, size_t ws_size,
                              hipStream_t stream) {
  (void)in_sizes; (void)n_in; (void)out_size; (void)ws_size;
  char* ws = (char*)d_ws;
  const size_t MB = 1024 * 1024;

  unsigned short* xb_re  = (unsigned short*)(ws + 0 * MB);   // [2048,1024] bf16
  unsigned short* xb_im  = (unsigned short*)(ws + 4 * MB);
  unsigned short* wqb_re = (unsigned short*)(ws + 8 * MB);   // [3072,1024] bf16
  unsigned short* wqb_im = (unsigned short*)(ws + 14 * MB);
  unsigned short* wob_re = (unsigned short*)(ws + 20 * MB);  // [1024,1024] bf16
  unsigned short* wob_im = (unsigned short*)(ws + 22 * MB);
  float*          qkv_r  = (float*)(ws + 24 * MB);           // [2048,3072] f32
  float*          qkv_i  = (float*)(ws + 48 * MB);
  unsigned short* Qr     = (unsigned short*)(ws + 72 * MB);  // [64,512,64] bf16
  unsigned short* Qi     = (unsigned short*)(ws + 76 * MB);
  unsigned short* Kr     = (unsigned short*)(ws + 80 * MB);
  unsigned short* Ki     = (unsigned short*)(ws + 84 * MB);
  unsigned short* Vtr    = (unsigned short*)(ws + 88 * MB);  // [64,64,512] bf16
  unsigned short* Vti    = (unsigned short*)(ws + 92 * MB);
  unsigned short* attn_r = (unsigned short*)(ws + 96 * MB);  // [2048,1024] bf16
  unsigned short* attn_i = (unsigned short*)(ws + 100 * MB);

  // 1) convert inputs to bf16
  cvt_f32_bf16<<<(2048 * 1024 + 255) / 256, 256, 0, stream>>>((const float*)d_in[0], xb_re, 2048 * 1024);
  cvt_f32_bf16<<<(2048 * 1024 + 255) / 256, 256, 0, stream>>>((const float*)d_in[1], xb_im, 2048 * 1024);
  cvt_f32_bf16<<<(3072 * 1024 + 255) / 256, 256, 0, stream>>>((const float*)d_in[2], wqb_re, 3072 * 1024);
  cvt_f32_bf16<<<(3072 * 1024 + 255) / 256, 256, 0, stream>>>((const float*)d_in[3], wqb_im, 3072 * 1024);
  cvt_f32_bf16<<<(1024 * 1024 + 255) / 256, 256, 0, stream>>>((const float*)d_in[4], wob_re, 1024 * 1024);
  cvt_f32_bf16<<<(1024 * 1024 + 255) / 256, 256, 0, stream>>>((const float*)d_in[5], wob_im, 1024 * 1024);

  // 2) QKV complex projection: [2048,1024] x [3072,1024]^T
  cgemm_cplx_bf16<<<dim3(3072 / TN, 2048 / TM), 256, 0, stream>>>(
      xb_re, xb_im, wqb_re, wqb_im, qkv_r, qkv_i, 2048, 3072, 1024);

  // 3) RoPE + head split (Q,K rotated; V transposed)
  rope_split_kernel<<<(64 * 512 * 64) / 256, 256, 0, stream>>>(
      qkv_r, qkv_i, Qr, Qi, Kr, Ki, Vtr, Vti);

  // 4) fused attention, 256 KB dynamic LDS per workgroup (320 KB WGP LDS)
  hipFuncSetAttribute(reinterpret_cast<const void*>(attn_kernel),
                      hipFuncAttributeMaxDynamicSharedMemorySize, 256 * 1024);
  attn_kernel<<<dim3(512 / 64, 16, 4), 256, 256 * 1024, stream>>>(
      Qr, Qi, Kr, Ki, Vtr, Vti, attn_r, attn_i);

  // 5) output complex projection straight into d_out: [y_r | y_i]
  float* yr = (float*)d_out;
  float* yi = yr + (size_t)2048 * 1024;
  cgemm_cplx_bf16<<<dim3(1024 / TN, 2048 / TM), 256, 0, stream>>>(
      attn_r, attn_i, wob_re, wob_im, yr, yi, 2048, 1024, 1024);
}
